// NestedEventMamba_63221918597618
// MI455X (gfx1250) — matrix operand order repaired and checked by reading, not compiled, not run on hardware
//
#include <hip/hip_runtime.h>
#include <hip/hip_bf16.h>

// ---------------- model constants ----------------
#define BB    8
#define WW    32
#define LL    256
#define FEATN 130
#define FEATP 132   // padded row stride for masked feats (16B-aligned rows)
#define DIMN  128
#define DIN   256
#define DSN   32

typedef __attribute__((ext_vector_type(16))) __bf16 v16bf;
typedef __attribute__((ext_vector_type(8)))  float  v8f;

enum { ACT_NONE = 0, ACT_GELU = 1, ACT_SILU = 2, ACT_SOFTPLUS = 3 };

__device__ __forceinline__ float gelu_f(float x) {
    return 0.5f * x * (1.0f + erff(x * 0.7071067811865475f));
}
__device__ __forceinline__ float silu_f(float x) {
    return x * (1.0f / (1.0f + __expf(-x)));
}
__device__ __forceinline__ float softplus_f(float x) {
    return (x > 20.0f) ? x : log1pf(__expf(x));
}
__device__ __forceinline__ float apply_act(float v, int act) {
    if (act == ACT_GELU) return gelu_f(v);
    if (act == ACT_SILU) return silu_f(v);
    if (act == ACT_SOFTPLUS) return softplus_f(v);
    return v;
}
__device__ __forceinline__ unsigned short f2bf(float f) {
    unsigned int u = __float_as_uint(f);
    u += 0x7fffu + ((u >> 16) & 1u);   // round-to-nearest-even
    return (unsigned short)(u >> 16);
}
// packed f32x2 -> bf16x2 via hardware v_cvt_pk_bf16_f32
__device__ __forceinline__ unsigned int pack2bf(float a, float b) {
    float2 f2; f2.x = a; f2.y = b;
    __hip_bfloat162 h = __float22bfloat162_rn(f2);
    unsigned int u;
    __builtin_memcpy(&u, &h, 4);
    return u;
}

// ---------------- WMMA GEMM: C = epilogue(A[M,K] @ W[N,K]^T) ----------------
// bf16 in LDS (converted from fp32 during staging), f32 acc via v_wmma_f32_16x16x32_bf16.
// 256 threads = 8 waves; block tile 128x64; wave tile 32x32 (4 accumulators);
// double-buffered LDS. FAST=true: all dims aligned -> zero bounds checks.
#define LDA_SH 40   // halfword row stride, A tile (80B rows: 16B aligned, bank stride 20)
#define LDB_SH 56   // halfword row stride, B tile (112B rows: 16B aligned, bank stride 28)

template <bool FAST>
__global__ __launch_bounds__(256) void k_gemm_t(
    const float* __restrict__ A, int lda,
    const float* __restrict__ W,               // [N,K] row-major
    const float* __restrict__ bias,            // [N] or null
    const float* __restrict__ res, int ldres,  // residual or null
    float* __restrict__ C, int ldc,
    int M, int N, int K, int act)
{
    __shared__ __align__(16) unsigned short At[2][128 * LDA_SH];
    __shared__ __align__(16) unsigned short Bt[2][64 * LDB_SH];

    const int tid  = threadIdx.x;
    const int lane = tid & 31;
    const int wid  = tid >> 5;
    const int m0 = blockIdx.y * 128;
    const int n0 = blockIdx.x * 64;
    const int wm = (wid & 3) * 32;
    const int wn = (wid >> 2) * 32;

    v8f c00 = {}, c01 = {}, c10 = {}, c11 = {};

    // fast staging: batch all global loads first (loads issue back-to-back,
    // single wait), then pack to bf16 (v_cvt_pk_bf16_f32) and ds_store_b64.
    auto stageA = [&](unsigned short* dst, int k0) {
        if constexpr (FAST) {
            float4 va[4];
#pragma unroll
            for (int r = 0; r < 4; ++r) {
                int m  = (tid >> 3) + r * 32;
                int k4 = (tid & 7) * 4;
                va[r] = *reinterpret_cast<const float4*>(A + (long)(m0 + m) * lda + (k0 + k4));
            }
#pragma unroll
            for (int r = 0; r < 4; ++r) {
                int m  = (tid >> 3) + r * 32;
                int k4 = (tid & 7) * 4;
                *reinterpret_cast<uint2*>(&dst[m * LDA_SH + k4]) =
                    make_uint2(pack2bf(va[r].x, va[r].y), pack2bf(va[r].z, va[r].w));
            }
        } else {
#pragma unroll
            for (int r = 0; r < 16; ++r) {
                int i = tid + r * 256;
                int m = i >> 5, k = i & 31;
                int gm = m0 + m, gk = k0 + k;
                float v = (gm < M && gk < K) ? A[(long)gm * lda + gk] : 0.0f;
                dst[m * LDA_SH + k] = f2bf(v);
            }
        }
    };
    auto stageB = [&](unsigned short* dst, int k0) {
        if constexpr (FAST) {
            float4 vb[2];
#pragma unroll
            for (int r = 0; r < 2; ++r) {
                int n  = (tid >> 3) + r * 32;
                int k4 = (tid & 7) * 4;
                vb[r] = *reinterpret_cast<const float4*>(W + (long)(n0 + n) * K + (k0 + k4));
            }
#pragma unroll
            for (int r = 0; r < 2; ++r) {
                int n  = (tid >> 3) + r * 32;
                int k4 = (tid & 7) * 4;
                *reinterpret_cast<uint2*>(&dst[n * LDB_SH + k4]) =
                    make_uint2(pack2bf(vb[r].x, vb[r].y), pack2bf(vb[r].z, vb[r].w));
            }
        } else {
#pragma unroll
            for (int r = 0; r < 8; ++r) {
                int i = tid + r * 256;
                int n = i >> 5, k = i & 31;
                int gn = n0 + n, gk = k0 + k;
                float v = (gn < N && gk < K) ? W[(long)gn * K + gk] : 0.0f;
                dst[n * LDB_SH + k] = f2bf(v);
            }
        }
    };

    // prologue
    stageA(At[0], 0);
    stageB(Bt[0], 0);

    int buf = 0;
#pragma unroll 1
    for (int k0 = 0; k0 < K; k0 += 32) {
        __syncthreads();
        int nk = k0 + 32;
        if (nk < K) {                     // stage next tile; overlaps with WMMA below
            stageA(At[buf ^ 1], nk);
            stageB(Bt[buf ^ 1], nk);
        }

        const unsigned short* Ab = At[buf];
        const unsigned short* Bb = Bt[buf];
        const int h  = lane >> 4;
        const int lm = lane & 15;
        union FragU { uint4 q[2]; v16bf v; };
        FragU fa0, fa1, fb0, fb1;
        // A fragments (ISA 7.12.2): per-lane K runs [8h,8h+8) and [16+8h,16+8h+8)
        const int ma0 = wm + lm, ma1 = wm + 16 + lm;
        fa0.q[0] = *reinterpret_cast<const uint4*>(&Ab[ma0 * LDA_SH + 8 * h]);
        fa0.q[1] = *reinterpret_cast<const uint4*>(&Ab[ma0 * LDA_SH + 16 + 8 * h]);
        fa1.q[0] = *reinterpret_cast<const uint4*>(&Ab[ma1 * LDA_SH + 8 * h]);
        fa1.q[1] = *reinterpret_cast<const uint4*>(&Ab[ma1 * LDA_SH + 16 + 8 * h]);
        // B fragments: per-lane K runs [8h,8h+8) and [16+8h,16+8h+8) (same as A)
        const int nb0 = wn + lm, nb1 = wn + 16 + lm;
        fb0.q[0] = *reinterpret_cast<const uint4*>(&Bb[nb0 * LDB_SH + 8 * h]);
        fb0.q[1] = *reinterpret_cast<const uint4*>(&Bb[nb0 * LDB_SH + 16 + 8 * h]);
        fb1.q[0] = *reinterpret_cast<const uint4*>(&Bb[nb1 * LDB_SH + 8 * h]);
        fb1.q[1] = *reinterpret_cast<const uint4*>(&Bb[nb1 * LDB_SH + 16 + 8 * h]);

        c00 = __builtin_amdgcn_wmma_f32_16x16x32_bf16(false, fa0.v, false, fb0.v, (short)0, c00, false, false);
        c01 = __builtin_amdgcn_wmma_f32_16x16x32_bf16(false, fa0.v, false, fb1.v, (short)0, c01, false, false);
        c10 = __builtin_amdgcn_wmma_f32_16x16x32_bf16(false, fa1.v, false, fb0.v, (short)0, c10, false, false);
        c11 = __builtin_amdgcn_wmma_f32_16x16x32_bf16(false, fa1.v, false, fb1.v, (short)0, c11, false, false);
        buf ^= 1;
    }

    // epilogue: D layout — VGPR r: lanes0-15 row=r, lanes16-31 row=r+8; col = lane%16
    const int h    = lane >> 4;
    const int lm   = lane & 15;
    const int colA = n0 + wn + lm;
    const int colB = colA + 16;
    const int r0   = m0 + wm + 8 * h;

    auto emit = [&](const v8f& cc, int rbase, int col) {
#pragma unroll
        for (int r = 0; r < 8; ++r) {
            int row = rbase + r;
            if constexpr (!FAST) { if (row >= M || col >= N) continue; }
            float v = cc[r];
            if (bias) v += bias[col];
            v = apply_act(v, act);
            if (res)  v += res[(long)row * ldres + col];
            C[(long)row * ldc + col] = v;
        }
    };
    emit(c00, r0,      colA);
    emit(c01, r0,      colB);
    emit(c10, r0 + 16, colA);
    emit(c11, r0 + 16, colB);
}

// ---------------- elementwise / norm / conv kernels ----------------

// masked feats into padded [T, FEATP] buffer (cols >=130 zero)
__global__ void k_mask(const float* __restrict__ feats, const int* __restrict__ seg,
                       float* __restrict__ o, long total) {
    long i = (long)blockIdx.x * blockDim.x + threadIdx.x;
    if (i >= total) return;
    int  f = (int)(i % FEATP);
    long t = i / FEATP;          // token index (b*W+w)*L + l
    float v = 0.0f;
    if (f < FEATN) {
        int l  = (int)(t % LL);
        int bw = (int)(t / LL);
        int s  = seg[bw]; if (s < 1) s = 1;
        if (l < s) v = feats[t * FEATN + f];
    }
    o[i] = v;
}

__global__ void k_add(const float* __restrict__ a, const float* __restrict__ b,
                      float* __restrict__ o, long n) {
    long i = (long)blockIdx.x * blockDim.x + threadIdx.x;
    if (i < n) o[i] = a[i] + b[i];
}

// wave-per-row RMSNorm over 128 features (wave32: 4 floats/lane)
__global__ __launch_bounds__(256) void k_rmsnorm(const float* __restrict__ x,
                                                 const float* __restrict__ w,
                                                 float* __restrict__ o, long T) {
    long row = (long)blockIdx.x * 8 + (threadIdx.x >> 5);
    int lane = threadIdx.x & 31;
    if (row >= T) return;
    float4 v = ((const float4*)(x + row * DIMN))[lane];
    float ss = v.x * v.x + v.y * v.y + v.z * v.z + v.w * v.w;
    for (int off = 16; off; off >>= 1) ss += __shfl_xor(ss, off, 32);
    float r = rsqrtf(ss * (1.0f / 128.0f) + 1.1920929e-07f);
    float4 wv = ((const float4*)w)[lane];
    float4 ov;
    ov.x = v.x * r * wv.x; ov.y = v.y * r * wv.y;
    ov.z = v.z * r * wv.z; ov.w = v.w * r * wv.w;
    ((float4*)(o + row * DIMN))[lane] = ov;
}

// wave-per-row LayerNorm over 128 features, optional fused SiLU
__global__ __launch_bounds__(256) void k_layernorm(const float* __restrict__ x,
                                                   const float* __restrict__ g,
                                                   const float* __restrict__ b,
                                                   float* __restrict__ o, long T,
                                                   int do_silu) {
    long row = (long)blockIdx.x * 8 + (threadIdx.x >> 5);
    int lane = threadIdx.x & 31;
    if (row >= T) return;
    float4 v = ((const float4*)(x + row * DIMN))[lane];
    float s1 = v.x + v.y + v.z + v.w;
    float s2 = v.x * v.x + v.y * v.y + v.z * v.z + v.w * v.w;
    for (int off = 16; off; off >>= 1) {
        s1 += __shfl_xor(s1, off, 32);
        s2 += __shfl_xor(s2, off, 32);
    }
    float m = s1 * (1.0f / 128.0f);
    float var = s2 * (1.0f / 128.0f) - m * m;
    float r = rsqrtf(var + 1e-5f);
    float4 gv = ((const float4*)g)[lane];
    float4 bv = ((const float4*)b)[lane];
    float4 ov;
    ov.x = (v.x - m) * r * gv.x + bv.x;
    ov.y = (v.y - m) * r * gv.y + bv.y;
    ov.z = (v.z - m) * r * gv.z + bv.z;
    ov.w = (v.w - m) * r * gv.w + bv.w;
    if (do_silu) {
        ov.x = silu_f(ov.x); ov.y = silu_f(ov.y);
        ov.z = silu_f(ov.z); ov.w = silu_f(ov.w);
    }
    ((float4*)(o + row * DIMN))[lane] = ov;
}

// depthwise 1-D conv (cross-correlation, matches torch/jax), optional fused SiLU
__global__ void k_dwconv(const float* __restrict__ x, int ldx,
                         const float* __restrict__ w, const float* __restrict__ b,
                         float* __restrict__ o, int ldo,
                         int Nn, int L, int D, int K, int padL, int act) {
    long i = (long)blockIdx.x * blockDim.x + threadIdx.x;
    long total = (long)Nn * L * D;
    if (i >= total) return;
    int d = (int)(i % D);
    long t = i / D;
    int l = (int)(t % L);
    int n = (int)(t / L);
    float acc = b[d];
    for (int k = 0; k < K; ++k) {
        int ls = l + k - padL;
        if (ls >= 0 && ls < L) acc += w[d * K + k] * x[((long)n * L + ls) * ldx + d];
    }
    if (act == ACT_SILU) acc = silu_f(acc);
    o[((long)n * L + l) * ldo + d] = acc;
}

// ---------------- selective scan (Mamba SSM) ----------------
__global__ __launch_bounds__(256) void k_scan(
    const float* __restrict__ dt, int lddt,
    const float* __restrict__ xm, int ldxm,
    const float* __restrict__ dbc, int lddbc,   // B at col 8, C at col 40
    const float* __restrict__ z, int ldz,
    const float* __restrict__ A_log,            // [256,32]
    const float* __restrict__ Dv,               // [256]
    float* __restrict__ y, int L)
{
    __shared__ float Ash[DIN * 33];
    __shared__ float Bs[8 * DSN];
    __shared__ float Cs[8 * DSN];
    const int n = blockIdx.x;
    const int d = threadIdx.x;

    for (int i = threadIdx.x; i < DIN * DSN; i += 256) {
        int dd = i >> 5, s = i & 31;
        Ash[dd * 33 + s] = -__expf(A_log[i]);
    }
    float h[DSN];
#pragma unroll
    for (int s = 0; s < DSN; ++s) h[s] = 0.0f;
    const float Dd = Dv[d];
    __syncthreads();

    for (int l0 = 0; l0 < L; l0 += 8) {
        __syncthreads();
        for (int i = threadIdx.x; i < 8 * 64; i += 256) {
            int ll = i >> 6, c = i & 63;
            long t = (long)n * L + l0 + ll;
            float v = dbc[t * lddbc + 8 + c];
            if (c < 32) Bs[ll * 32 + c] = v;
            else        Cs[ll * 32 + (c - 32)] = v;
        }
        __syncthreads();
#pragma unroll 1
        for (int ll = 0; ll < 8; ++ll) {
            long t = (long)n * L + l0 + ll;
            float dtv = dt[t * lddt + d];
            float xv  = xm[t * ldxm + d];
            float dx  = dtv * xv;
            float acc = 0.0f;
#pragma unroll
            for (int s = 0; s < DSN; ++s) {
                float hs = h[s] * __expf(dtv * Ash[d * 33 + s]) + dx * Bs[ll * 32 + s];
                h[s] = hs;
                acc += hs * Cs[ll * 32 + s];
            }
            float zv = z[t * ldz + d];
            y[t * DIN + d] = (acc + xv * Dd) * silu_f(zv);
        }
    }
}

// ---------------- pooling / head ----------------

__global__ __launch_bounds__(128) void k_pool(const float* __restrict__ x,
                                              const int* __restrict__ seg,
                                              float* __restrict__ win, int L) {
    int bw = blockIdx.x;
    int d  = threadIdx.x;
    int s = seg[bw]; if (s < 1) s = 1;
    float acc = 0.0f, mx = -__builtin_inff();
    for (int l = 0; l < s; ++l) {
        float v = x[((long)bw * L + l) * DIMN + d];
        acc += v;
        mx = fmaxf(mx, v);
    }
    win[(long)bw * DIMN + d] = 0.5f * (acc / (float)s + mx);
}

__global__ void k_seqmask(const float* __restrict__ win, const int* __restrict__ seql,
                          float* __restrict__ seq, long total) {
    long i = (long)blockIdx.x * blockDim.x + threadIdx.x;
    if (i >= total) return;
    long bw = i / DIMN;
    int w = (int)(bw % WW), b = (int)(bw / WW);
    int s = seql[b]; if (s < 1) s = 1;
    seq[i] = (w < s) ? win[i] : 0.0f;
}

__global__ __launch_bounds__(128) void k_attnpool(const float* __restrict__ seq,
                                                  const int* __restrict__ seql,
                                                  const float* __restrict__ pw,
                                                  const float* __restrict__ pb,
                                                  float* __restrict__ g) {
    __shared__ float wts[WW];
    const int b = blockIdx.x;
    const int tid = threadIdx.x;
    int s = seql[b]; if (s < 1) s = 1;
    if (tid < WW) {
        float al = pb[0];
        for (int dd = 0; dd < DIMN; ++dd)
            al += seq[((long)b * WW + tid) * DIMN + dd] * pw[dd];
        float v = (tid < s) ? al : -__builtin_inff();
        float mx = v;
        for (int off = 16; off; off >>= 1) mx = fmaxf(mx, __shfl_xor(mx, off, 32));
        float e = (tid < s) ? __expf(v - mx) : 0.0f;
        float sum = e;
        for (int off = 16; off; off >>= 1) sum += __shfl_xor(sum, off, 32);
        wts[tid] = e / sum;
    }
    __syncthreads();
    float acc = 0.0f;
    for (int w = 0; w < WW; ++w)
        acc += seq[((long)b * WW + w) * DIMN + tid] * wts[w];
    g[(long)b * DIMN + tid] = acc;
}

// ---------------- host-side orchestration ----------------

static void launch_gemm(hipStream_t s, const float* A, int lda, const float* W,
                        const float* bias, const float* res, int ldres,
                        float* C, int ldc, long M, int N, int K, int act) {
    dim3 g((unsigned)((N + 63) / 64), (unsigned)((M + 127) / 128));
    bool fast = ((M & 127) == 0) && ((N & 63) == 0) && ((K & 31) == 0) && ((lda & 3) == 0);
    if (fast)
        k_gemm_t<true><<<g, 256, 0, s>>>(A, lda, W, bias, res, ldres, C, ldc, (int)M, N, K, act);
    else
        k_gemm_t<false><<<g, 256, 0, s>>>(A, lda, W, bias, res, ldres, C, ldc, (int)M, N, K, act);
}

// block params (23): 0 norm1_w, 1 conv_w, 2 conv_b, 3 lnc_g, 4 lnc_b, 5 in_proj,
// 6 sconv_w, 7 sconv_b, 8 x_proj, 9 dt_w, 10 dt_b, 11 A_log, 12 D, 13 out_proj,
// 14 post_w, 15 post_b, 16 lnp_g, 17 lnp_b, 18 norm2_w, 19 mlp_w1, 20 mlp_b1,
// 21 mlp_w2, 22 mlp_b2
static void run_block(hipStream_t s, void* const* d_in, int pb, float* Xb,
                      int Nseq, int L, int Kc,
                      float* XN, float* XC, float* XZ, float* XM, float* DBC, float* Y) {
    auto P = [&](int i) { return (const float*)d_in[i]; };
    long T = (long)Nseq * L;
    int nbRow = (int)((T + 7) / 8);
    long tD = T * DIMN, tDI = T * DIN;

    k_rmsnorm<<<nbRow, 256, 0, s>>>(Xb, P(pb + 0), XN, T);
    k_dwconv<<<(unsigned)((tD + 255) / 256), 256, 0, s>>>(XN, DIMN, P(pb + 1), P(pb + 2),
                                                          XC, DIMN, Nseq, L, DIMN, Kc, Kc / 2, ACT_NONE);
    k_layernorm<<<nbRow, 256, 0, s>>>(XC, P(pb + 3), P(pb + 4), XC, T, 1);
    k_add<<<(unsigned)((tD + 255) / 256), 256, 0, s>>>(XN, XC, XN, tD);   // mamba input
    // mamba
    launch_gemm(s, XN, DIMN, P(pb + 5), nullptr, nullptr, 0, XZ, 512, T, 512, DIMN, ACT_NONE);
    k_dwconv<<<(unsigned)((tDI + 255) / 256), 256, 0, s>>>(XZ, 512, P(pb + 6), P(pb + 7),
                                                           XM, DIN, Nseq, L, DIN, 4, 3, ACT_SILU);
    launch_gemm(s, XM, DIN, P(pb + 8), nullptr, nullptr, 0, DBC, 80, T, 72, DIN, ACT_NONE);
    launch_gemm(s, DBC, 80, P(pb + 9), P(pb + 10), nullptr, 0, XZ, 512, T, DIN, 8, ACT_SOFTPLUS);
    k_scan<<<Nseq, 256, 0, s>>>(XZ, 512, XM, DIN, DBC, 80, XZ + 256, 512,
                                P(pb + 11), P(pb + 12), Y, L);
    launch_gemm(s, Y, DIN, P(pb + 13), nullptr, nullptr, 0, XC, DIMN, T, DIMN, DIN, ACT_NONE);
    // post conv + LN + silu + residual
    k_dwconv<<<(unsigned)((tD + 255) / 256), 256, 0, s>>>(XC, DIMN, P(pb + 14), P(pb + 15),
                                                          XN, DIMN, Nseq, L, DIMN, 3, 1, ACT_NONE);
    k_layernorm<<<nbRow, 256, 0, s>>>(XN, P(pb + 16), P(pb + 17), XN, T, 1);
    k_add<<<(unsigned)((tD + 255) / 256), 256, 0, s>>>(Xb, XN, Xb, tD);
    // MLP with fused GELU + residual
    k_rmsnorm<<<nbRow, 256, 0, s>>>(Xb, P(pb + 18), XN, T);
    launch_gemm(s, XN, DIMN, P(pb + 19), P(pb + 20), nullptr, 0, XZ, 512, T, 512, DIMN, ACT_GELU);
    launch_gemm(s, XZ, 512, P(pb + 21), P(pb + 22), Xb, DIMN, Xb, DIMN, T, DIMN, 512, ACT_NONE);
}

extern "C" void kernel_launch(void* const* d_in, const int* in_sizes, int n_in,
                              void* d_out, int out_size, void* d_ws, size_t ws_size,
                              hipStream_t stream) {
    (void)in_sizes; (void)n_in; (void)out_size; (void)ws_size;
    const float* feats = (const float*)d_in[0];
    const int*   seg   = (const int*)d_in[1];
    const int*   seql  = (const int*)d_in[2];
    auto P = [&](int i) { return (const float*)d_in[i]; };

    // input index map: 3..6 adapter; 7 intra0; 30 intra1; 53 inter0; 76 inter1;
    // 99 pool_w; 100 pool_b; 101..106 head
    const size_t T = (size_t)BB * WW * LL;   // 65536 intra tokens
    float* X   = (float*)d_ws;               // [T,128] residual stream
    float* XN  = X   + T * 128;
    float* XC  = XN  + T * 128;
    float* XZ  = XC  + T * 128;              // [T,512] (also masked feats [T,132])
    float* XM  = XZ  + T * 512;              // [T,256]
    float* DBC = XM  + T * 256;              // [T,80] (72 used)
    float* Y   = DBC + T * 80;               // [T,256]
    float* SEQ = Y   + T * 256;              // [256,128]
    float* WIN = SEQ + (size_t)BB * WW * 128;
    float* G   = WIN + (size_t)BB * WW * 128;
    float* H1  = G   + (size_t)BB * 128;

    // mask + adapter
    {
        long tot = (long)T * FEATP;
        k_mask<<<(unsigned)((tot + 255) / 256), 256, 0, stream>>>(feats, seg, XZ, tot);
    }
    launch_gemm(stream, XZ, FEATP, P(3), P(4), nullptr, 0, XN, DIMN, (long)T, DIMN, FEATN, ACT_GELU);
    launch_gemm(stream, XN, DIMN,  P(5), P(6), nullptr, 0, X,  DIMN, (long)T, DIMN, DIMN, ACT_NONE);

    // intra blocks (256 sequences of length 256, conv K=11)
    run_block(stream, d_in, 7,  X, BB * WW, LL, 11, XN, XC, XZ, XM, DBC, Y);
    run_block(stream, d_in, 30, X, BB * WW, LL, 11, XN, XC, XZ, XM, DBC, Y);

    // window pooling + sequence mask
    k_pool<<<BB * WW, 128, 0, stream>>>(X, seg, WIN, LL);
    {
        long tot = (long)BB * WW * DIMN;
        k_seqmask<<<(unsigned)((tot + 255) / 256), 256, 0, stream>>>(WIN, seql, SEQ, tot);
    }

    // inter blocks (8 sequences of length 32, conv K=3)
    run_block(stream, d_in, 53, SEQ, BB, WW, 3, XN, XC, XZ, XM, DBC, Y);
    run_block(stream, d_in, 76, SEQ, BB, WW, 3, XN, XC, XZ, XM, DBC, Y);

    // attention pooling + head
    k_attnpool<<<BB, 128, 0, stream>>>(SEQ, seql, P(99), P(100), G);
    k_layernorm<<<1, 256, 0, stream>>>(G, P(101), P(102), G, BB, 0);
    launch_gemm(stream, G,  DIMN, P(103), P(104), nullptr, 0, H1, 64, BB, 64, DIMN, ACT_GELU);
    launch_gemm(stream, H1, 64,   P(105), P(106), nullptr, 0, (float*)d_out, 11, BB, 11, 64, ACT_NONE);
}
